// TRTTN_38955353375094
// MI455X (gfx1250) — compile-verified, matrix operand order checked by reading, not run on hardware
//
#include <hip/hip_runtime.h>
#include <hip/hip_bf16.h>
#include <stdint.h>

// ---------------- types ----------------
typedef __bf16 bf16_t;
typedef __attribute__((ext_vector_type(16))) __bf16 v16bf;
typedef __attribute__((ext_vector_type(8)))  __bf16 v8bf;
typedef __attribute__((ext_vector_type(8)))  float  v8f;

#define N_B   16
#define SEQ   20
#define EMB   300
#define VD    256
#define P_PIX 1024              // 32*32
#define M_ROWS (N_B * P_PIX)    // 16384
#define PADW  34                // 32 + halo

// 16x32 bf16 A/B fragment row-chunk per CDNA5 16-bit layout:
// lanes 0-15 hold K {0..7,16..23}, lanes 16-31 hold K {8..15,24..31}.
__device__ __forceinline__ v16bf load_frag(const bf16_t* base, int lo) {
    v8bf c0 = *reinterpret_cast<const v8bf*>(base + lo);
    v8bf c1 = *reinterpret_cast<const v8bf*>(base + lo + 16);
    v16bf r;
#pragma unroll
    for (int i = 0; i < 8; ++i) { r[i] = c0[i]; r[i + 8] = c1[i]; }
    return r;
}

// ---------------- one-time setup kernels ----------------

__global__ void cvt_kernel(const float* __restrict__ src, bf16_t* __restrict__ dst, int n) {
    int i = blockIdx.x * 256 + threadIdx.x;
    if (i < n) dst[i] = (bf16_t)src[i];
}

// wp[tap][oc][ic] = mconv_w[oc][ic][ky][kx], ic<256 (dynamic channels)
__global__ void wpack_kernel(const float* __restrict__ mw, bf16_t* __restrict__ wp) {
    size_t i = (size_t)blockIdx.x * 256 + threadIdx.x;   // 9*256*256
    int ic  = i & 255;
    int oc  = (i >> 8) & 255;
    int tap = (int)(i >> 16);
    int ky = tap / 3, kx = tap % 3;
    wp[i] = (bf16_t)mw[(((size_t)oc * 776 + ic) * 3 + ky) * 3 + kx];
}

// Static conv contribution of channels 256..775 (8 spatial + 512 hn-broadcast).
// S layout: [(n*1024+p)*256 + oc].
__global__ void static_kernel(const float* __restrict__ mw, const float* __restrict__ hn,
                              float* __restrict__ S) {
    size_t idx = (size_t)blockIdx.x * 256 + threadIdx.x;  // 16*1024*256
    int oc = idx & 255;
    int p  = (int)((idx >> 8) & 1023);
    int n  = (int)(idx >> 18);
    int y = p >> 5, x = p & 31;
    float acc = 0.f;
    for (int dy = -1; dy <= 1; ++dy) {
        int yy = y + dy; if (yy < 0 || yy >= 32) continue;
        for (int dx = -1; dx <= 1; ++dx) {
            int xx = x + dx; if (xx < 0 || xx >= 32) continue;
            int ky = dy + 1, kx = dx + 1;
            const float* wbase = mw + (((size_t)oc * 776 + 256) * 3 + ky) * 3 + kx;
            float xmin = xx * (2.f / 32.f) - 1.f, xmax = (xx + 1) * (2.f / 32.f) - 1.f;
            float ymin = yy * (2.f / 32.f) - 1.f, ymax = (yy + 1) * (2.f / 32.f) - 1.f;
            float sp[8] = { xmin, ymin, xmax, ymax,
                            0.5f * (xmin + xmax), 0.5f * (ymin + ymax),
                            1.f / 32.f, 1.f / 32.f };
            for (int c = 0; c < 8; ++c) acc += wbase[(size_t)c * 9] * sp[c];
            const float* wh = wbase + (size_t)8 * 9;
            const float* hv = hn + (size_t)n * 512;
            for (int c = 0; c < 512; ++c) acc += wh[(size_t)c * 9] * hv[c];
        }
    }
    S[idx] = acc;
}

// plain NCHW f32 copy into d_out
__global__ void init_out_kernel(const float* __restrict__ feature, float* __restrict__ out_nchw) {
    size_t i = (size_t)blockIdx.x * 256 + threadIdx.x;
    out_nchw[i] = feature[i];
}

// zero-halo padded NHWC bf16 feature: featP[n][yy][xx][c], yy/xx in [0,34)
__global__ void init_pad_kernel(const float* __restrict__ feature, bf16_t* __restrict__ featP) {
    size_t i = (size_t)blockIdx.x * 256 + threadIdx.x;    // 16*34*34*256
    int c = i & 255;
    int pix = (int)(i >> 8);                              // 0..18495
    int xx = pix % PADW;
    int t  = pix / PADW;
    int yy = t % PADW;
    int n  = t / PADW;
    float v = 0.f;
    if (xx >= 1 && xx <= 32 && yy >= 1 && yy <= 32)
        v = feature[((size_t)n * 256 + c) * P_PIX + (yy - 1) * 32 + (xx - 1)];
    featP[i] = (bf16_t)v;
}

// ---------------- per-step kernels ----------------

// q = relu(emb @ qconv_w^T + qb); qproj = q @ Wq^T + bq   (16x256, tiny)
__global__ void q_kernel(const float* __restrict__ emb, const float* __restrict__ qw,
                         const float* __restrict__ qb, const float* __restrict__ wq,
                         const float* __restrict__ bq, bf16_t* __restrict__ qproj, int s) {
    __shared__ float qs[256];
    int n = blockIdx.x, c = threadIdx.x;
    const float* e = emb + ((size_t)n * SEQ + s) * EMB;
    float acc = qb[c];
    for (int k = 0; k < EMB; ++k) acc += qw[(size_t)c * EMB + k] * e[k];
    qs[c] = fmaxf(acc, 0.f);
    __syncthreads();
    float a2 = bq[c];
    for (int k = 0; k < 256; ++k) a2 += wq[(size_t)c * 256 + k] * qs[k];
    qproj[(size_t)n * 256 + c] = (bf16_t)a2;
}

// Implicit-GEMM 3x3 conv (zero-halo input => no bounds checks).
// Wave tile: 16 rows (same n,y; x0..x0+15) x 64 oc (4 accumulators).
__global__ void conv_gemm_kernel(const bf16_t* __restrict__ featP, const bf16_t* __restrict__ wpack,
                                 const float* __restrict__ S, float* __restrict__ out) {
    int wave = threadIdx.x >> 5, lane = threadIdx.x & 31;
    int task = blockIdx.x * 8 + wave;            // 4096 wave tasks
    int ng = task & 3;                           // oc group of 64
    int mt = task >> 2;                          // (n, y, xt)
    int xt = mt & 1, y = (mt >> 1) & 31, n = mt >> 6;
    int x0 = xt * 16;
    int Nbase = ng * 64;
    int col = lane & 15;
    int lo  = (lane & 16) ? 8 : 0;
    int rof = (lane & 16) ? 8 : 0;
    int m   = lane & 15;

    v8f c0, c1, c2, c3;
#pragma unroll
    for (int r = 0; r < 8; ++r) {
        size_t row = ((size_t)n * P_PIX + y * 32 + x0 + r + rof) * 256 + Nbase + col;
        c0[r] = S[row];
        c1[r] = S[row + 16];
        c2[r] = S[row + 32];
        c3[r] = S[row + 48];
    }

    for (int tap = 0; tap < 9; ++tap) {          // rolled: limits code size
        int ty = tap / 3, tx = tap % 3;          // dy+1, dx+1 (pad origin)
        const bf16_t* arow = featP + (((size_t)n * PADW + (y + ty)) * PADW + (x0 + m + tx)) * 256;
        const bf16_t* bb   = wpack + ((size_t)tap * 256 + Nbase + col) * 256;
#pragma unroll
        for (int kb = 0; kb < 256; kb += 32) {
            v16bf a  = load_frag(arow + kb, lo);
            v16bf b0 = load_frag(bb + kb, lo);
            c0 = __builtin_amdgcn_wmma_f32_16x16x32_bf16(false, a, false, b0, (short)0, c0, false, false);
            v16bf b1 = load_frag(bb + (size_t)16 * 256 + kb, lo);
            c1 = __builtin_amdgcn_wmma_f32_16x16x32_bf16(false, a, false, b1, (short)0, c1, false, false);
            v16bf b2 = load_frag(bb + (size_t)32 * 256 + kb, lo);
            c2 = __builtin_amdgcn_wmma_f32_16x16x32_bf16(false, a, false, b2, (short)0, c2, false, false);
            v16bf b3 = load_frag(bb + (size_t)48 * 256 + kb, lo);
            c3 = __builtin_amdgcn_wmma_f32_16x16x32_bf16(false, a, false, b3, (short)0, c3, false, false);
        }
    }
#pragma unroll
    for (int r = 0; r < 8; ++r) {
        size_t row = ((size_t)n * P_PIX + y * 32 + x0 + r + rof) * 256 + Nbase + col;
        out[row]      = fmaxf(c0[r], 0.f);       // relu before layernorm
        out[row + 16] = fmaxf(c1[r], 0.f);
        out[row + 32] = fmaxf(c2[r], 0.f);
        out[row + 48] = fmaxf(c3[r], 0.f);
    }
}

// Generic GEMM: C[16384,256] = A_bf16 * W[256,256]^T_bf16 + bias.
// Block = 8 M-tiles x 64 N-cols; weights staged to LDS via async-to-LDS DMA.
__global__ void gemm_kernel(const bf16_t* __restrict__ A, const bf16_t* __restrict__ W,
                            const float* __restrict__ bias, float* __restrict__ outF,
                            bf16_t* __restrict__ outB, int do_relu) {
    __shared__ bf16_t lds_w[64 * 256];           // 32 KB: 64 weight rows
    int t = threadIdx.x;
    int ngroup = blockIdx.x & 3;
    int mgroup = blockIdx.x >> 2;                // 0..127
    int Nbase = ngroup * 64;

    // stage 64 weight rows (32KB) with async global->LDS copies
    {
        const bf16_t* wsrc = W + (size_t)Nbase * 256;
        uint64_t gbase = (uint64_t)(uintptr_t)wsrc;
        uint32_t lbase = (uint32_t)(uintptr_t)(&lds_w[0]);
#pragma unroll
        for (int j = 0; j < 8; ++j) {
            uint32_t cidx = (uint32_t)t * 8u + j;        // 0..2047 16B chunks
            uint64_t ga = gbase + (uint64_t)cidx * 16u;
            uint32_t la = lbase + cidx * 16u;
            asm volatile("global_load_async_to_lds_b128 %0, %1, off"
                         :: "v"(la), "v"(ga) : "memory");
        }
        asm volatile("s_wait_asynccnt 0" ::: "memory");
    }
    __syncthreads();

    int wave = t >> 5, lane = t & 31;
    int Mbase = (mgroup * 8 + wave) * 16;
    int col = lane & 15;
    int lo  = (lane & 16) ? 8 : 0;
    int rof = (lane & 16) ? 8 : 0;
    int m   = lane & 15;

    float b0v = bias ? bias[Nbase + col]      : 0.f;
    float b1v = bias ? bias[Nbase + 16 + col] : 0.f;
    float b2v = bias ? bias[Nbase + 32 + col] : 0.f;
    float b3v = bias ? bias[Nbase + 48 + col] : 0.f;
    v8f c0, c1, c2, c3;
#pragma unroll
    for (int r = 0; r < 8; ++r) { c0[r] = b0v; c1[r] = b1v; c2[r] = b2v; c3[r] = b3v; }

    const bf16_t* arow = A + ((size_t)Mbase + m) * 256;
    const bf16_t* wl   = &lds_w[(size_t)col * 256];
#pragma unroll
    for (int kb = 0; kb < 256; kb += 32) {
        v16bf a  = load_frag(arow + kb, lo);
        v16bf b0 = load_frag(wl + kb, lo);
        c0 = __builtin_amdgcn_wmma_f32_16x16x32_bf16(false, a, false, b0, (short)0, c0, false, false);
        v16bf b1 = load_frag(wl + (size_t)16 * 256 + kb, lo);
        c1 = __builtin_amdgcn_wmma_f32_16x16x32_bf16(false, a, false, b1, (short)0, c1, false, false);
        v16bf b2 = load_frag(wl + (size_t)32 * 256 + kb, lo);
        c2 = __builtin_amdgcn_wmma_f32_16x16x32_bf16(false, a, false, b2, (short)0, c2, false, false);
        v16bf b3 = load_frag(wl + (size_t)48 * 256 + kb, lo);
        c3 = __builtin_amdgcn_wmma_f32_16x16x32_bf16(false, a, false, b3, (short)0, c3, false, false);
    }
#pragma unroll
    for (int r = 0; r < 8; ++r) {
        size_t row = ((size_t)Mbase + r + rof) * 256 + Nbase + col;
        float v0 = c0[r], v1 = c1[r], v2 = c2[r], v3 = c3[r];
        if (do_relu) { v0 = fmaxf(v0, 0.f); v1 = fmaxf(v1, 0.f); v2 = fmaxf(v2, 0.f); v3 = fmaxf(v3, 0.f); }
        if (outF) { outF[row] = v0; outF[row + 16] = v1; outF[row + 32] = v2; outF[row + 48] = v3; }
        if (outB) { outB[row] = (bf16_t)v0; outB[row + 16] = (bf16_t)v1;
                    outB[row + 32] = (bf16_t)v2; outB[row + 48] = (bf16_t)v3; }
    }
}

// Batch-attention per (p, h): scores(16x16) via 4 WMMA, register softmax over m,
// attn*V on VALU with 16B vector loads/stores (tiny FLOPs).
__global__ void attn_kernel(const bf16_t* __restrict__ qp, const bf16_t* __restrict__ kh,
                            const bf16_t* __restrict__ vv, bf16_t* __restrict__ obuf) {
    __shared__ float attn_lds[8][16][16];
    int wave = threadIdx.x >> 5, lane = threadIdx.x & 31;
    int task = blockIdx.x * 8 + wave;     // 2048 = P * NHEAD
    int p = task >> 1;
    int h = task & 1;
    int col = lane & 15;
    int lo  = (lane & 16) ? 8 : 0;
    int rof = (lane & 16) ? 8 : 0;

    v8f c;
#pragma unroll
    for (int r = 0; r < 8; ++r) c[r] = 0.f;
#pragma unroll
    for (int kb = 0; kb < 128; kb += 32) {
        const bf16_t* arow = qp + (size_t)(lane & 15) * 256 + h * 128 + kb;
        const bf16_t* brow = kh + ((size_t)(lane & 15) * P_PIX + p) * 256 + h * 128 + kb;
        v16bf a = load_frag(arow, lo);
        v16bf b = load_frag(brow, lo);
        c = __builtin_amdgcn_wmma_f32_16x16x32_bf16(false, a, false, b, (short)0, c, false, false);
    }
    const float scale = 0.088388347648318447f;   // 1/sqrt(128)
#pragma unroll
    for (int r = 0; r < 8; ++r) {
        float v = c[r] * scale;
        float mx = v;
        for (int off = 1; off < 16; off <<= 1) mx = fmaxf(mx, __shfl_xor(mx, off, 16));
        float e = __expf(v - mx);
        float s = e;
        for (int off = 1; off < 16; off <<= 1) s += __shfl_xor(s, off, 16);
        attn_lds[wave][r + rof][col] = e / s;
    }
    __syncthreads();
    // o[l, p, h*128+d] = sum_m attn[l][m] * V[m, p, h*128+d]
    int l  = lane >> 1;
    int dh = (lane & 1) * 64;
    float w[16];
#pragma unroll
    for (int mm = 0; mm < 16; ++mm) w[mm] = attn_lds[wave][l][mm];
    for (int dc = 0; dc < 64; dc += 16) {
        float a16[16];
#pragma unroll
        for (int i = 0; i < 16; ++i) a16[i] = 0.f;
        for (int mm = 0; mm < 16; ++mm) {
            const bf16_t* vp = vv + ((size_t)mm * P_PIX + p) * 256 + h * 128 + dh + dc;
            v8bf v0 = *reinterpret_cast<const v8bf*>(vp);
            v8bf v1 = *reinterpret_cast<const v8bf*>(vp + 8);
#pragma unroll
            for (int i = 0; i < 8; ++i) {
                a16[i]     += w[mm] * (float)v0[i];
                a16[8 + i] += w[mm] * (float)v1[i];
            }
        }
        bf16_t* op = obuf + ((size_t)l * P_PIX + p) * 256 + h * 128 + dh + dc;
        v8bf o0, o1;
#pragma unroll
        for (int i = 0; i < 8; ++i) { o0[i] = (bf16_t)a16[i]; o1[i] = (bf16_t)a16[8 + i]; }
        *reinterpret_cast<v8bf*>(op)     = o0;
        *reinterpret_cast<v8bf*>(op + 8) = o1;
    }
}

// LayerNorm over last dim (256), optional residual, f32 and/or bf16 outputs.
__global__ void ln_kernel(const float* __restrict__ in, const float* __restrict__ res,
                          const float* __restrict__ g, const float* __restrict__ b,
                          float* __restrict__ outF, bf16_t* __restrict__ outB) {
    int wave = threadIdx.x >> 5, lane = threadIdx.x & 31;
    size_t row = (size_t)blockIdx.x * 8 + wave;
    const float* x = in + row * 256;
    const float* rr = res ? res + row * 256 : nullptr;
    float v[8], s = 0.f, s2 = 0.f;
#pragma unroll
    for (int i = 0; i < 8; ++i) {
        int cidx = i * 32 + lane;
        float tt = x[cidx];
        if (rr) tt += rr[cidx];
        v[i] = tt; s += tt; s2 += tt * tt;
    }
    for (int off = 1; off < 32; off <<= 1) { s += __shfl_xor(s, off); s2 += __shfl_xor(s2, off); }
    float mean = s * (1.f / 256.f);
    float var  = s2 * (1.f / 256.f) - mean * mean;
    float inv  = rsqrtf(var + 1e-5f);
#pragma unroll
    for (int i = 0; i < 8; ++i) {
        int cidx = i * 32 + lane;
        float y = (v[i] - mean) * inv * g[cidx] + b[cidx];
        if (outF) outF[row * 256 + cidx] = y;
        if (outB) outB[row * 256 + cidx] = (bf16_t)y;
    }
}

// Conditional write-back: if words[s]!=0 commit (NCHW f32 + padded NHWC bf16)
__global__ void wb_kernel(const float* __restrict__ fea2, const int* __restrict__ words, int s,
                          float* __restrict__ out_nchw, bf16_t* __restrict__ featP) {
    if (words[s] == 0) return;
    size_t i = (size_t)blockIdx.x * 256 + threadIdx.x;   // (n,p,c)
    int c = i & 255;
    int p = (int)((i >> 8) & 1023);
    int n = (int)(i >> 18);
    int y = p >> 5, x = p & 31;
    float v = fea2[i];
    featP[(((size_t)n * PADW + (y + 1)) * PADW + (x + 1)) * 256 + c] = (bf16_t)v;
    out_nchw[((size_t)n * 256 + c) * P_PIX + p] = v;
}

// ---------------- host ----------------

extern "C" void kernel_launch(void* const* d_in, const int* in_sizes, int n_in,
                              void* d_out, int out_size, void* d_ws, size_t ws_size,
                              hipStream_t stream) {
    (void)in_sizes; (void)n_in; (void)out_size; (void)ws_size;
    const float* hn        = (const float*)d_in[1];
    const float* feature   = (const float*)d_in[2];
    const float* embedding = (const float*)d_in[3];
    const int*   words     = (const int*)  d_in[4];
    const float* qconv_w   = (const float*)d_in[5];
    const float* qconv_b   = (const float*)d_in[6];
    const float* mconv_w   = (const float*)d_in[7];
    const float* mnorm_g   = (const float*)d_in[8];
    const float* mnorm_b   = (const float*)d_in[9];
    const float* in_proj_w = (const float*)d_in[10];
    const float* in_proj_b = (const float*)d_in[11];
    const float* out_proj_w= (const float*)d_in[12];
    const float* out_proj_b= (const float*)d_in[13];
    const float* norm_g    = (const float*)d_in[14];
    const float* norm_b    = (const float*)d_in[15];
    const float* lin1_w    = (const float*)d_in[16];
    const float* lin1_b    = (const float*)d_in[17];
    const float* lin2_w    = (const float*)d_in[18];
    const float* lin2_b    = (const float*)d_in[19];
    const float* normf_g   = (const float*)d_in[20];
    const float* normf_b   = (const float*)d_in[21];

    uint8_t* ws = (uint8_t*)d_ws;
    size_t off = 0;
    auto alloc = [&](size_t bytes) -> void* {
        void* p = ws + off;
        off = (off + bytes + 255) & ~(size_t)255;
        return p;
    };
    const size_t F32 = (size_t)M_ROWS * 256 * sizeof(float);    // 16 MB
    const size_t BF  = (size_t)M_ROWS * 256 * sizeof(bf16_t);   // 8 MB
    const size_t PADN = (size_t)N_B * PADW * PADW * 256;        // padded elems

    float*  S       = (float*) alloc(F32);
    float*  acc     = (float*) alloc(F32);
    float*  t_f32   = (float*) alloc(F32);   // also reused as fea2
    float*  fea_f32 = (float*) alloc(F32);
    bf16_t* featP   = (bf16_t*)alloc(PADN * sizeof(bf16_t));
    bf16_t* t_bf    = (bf16_t*)alloc(BF);
    bf16_t* kh      = (bf16_t*)alloc(BF);
    bf16_t* vv      = (bf16_t*)alloc(BF);
    bf16_t* o_bf    = (bf16_t*)alloc(BF);
    bf16_t* fea_bf  = (bf16_t*)alloc(BF);
    bf16_t* ffa     = (bf16_t*)alloc(BF);
    bf16_t* wk_bf   = (bf16_t*)alloc(65536 * sizeof(bf16_t));
    bf16_t* wv_bf   = (bf16_t*)alloc(65536 * sizeof(bf16_t));
    bf16_t* wo_bf   = (bf16_t*)alloc(65536 * sizeof(bf16_t));
    bf16_t* l1_bf   = (bf16_t*)alloc(65536 * sizeof(bf16_t));
    bf16_t* l2_bf   = (bf16_t*)alloc(65536 * sizeof(bf16_t));
    bf16_t* wpack   = (bf16_t*)alloc((size_t)9 * 65536 * sizeof(bf16_t));
    bf16_t* qproj   = (bf16_t*)alloc((size_t)N_B * 256 * sizeof(bf16_t));

    // ---- one-time setup ----
    cvt_kernel<<<256, 256, 0, stream>>>(in_proj_w + 256 * 256, wk_bf, 65536);
    cvt_kernel<<<256, 256, 0, stream>>>(in_proj_w + 512 * 256, wv_bf, 65536);
    cvt_kernel<<<256, 256, 0, stream>>>(out_proj_w, wo_bf, 65536);
    cvt_kernel<<<256, 256, 0, stream>>>(lin1_w, l1_bf, 65536);
    cvt_kernel<<<256, 256, 0, stream>>>(lin2_w, l2_bf, 65536);
    wpack_kernel<<<2304, 256, 0, stream>>>(mconv_w, wpack);
    static_kernel<<<16384, 256, 0, stream>>>(mconv_w, hn, S);
    init_out_kernel<<<16384, 256, 0, stream>>>(feature, (float*)d_out);
    init_pad_kernel<<<18496, 256, 0, stream>>>(feature, featP);

    // ---- recurrent steps ----
    for (int s = 0; s < SEQ; ++s) {
        q_kernel<<<N_B, 256, 0, stream>>>(embedding, qconv_w, qconv_b,
                                          in_proj_w, in_proj_b, qproj, s);
        conv_gemm_kernel<<<512, 256, 0, stream>>>(featP, wpack, S, acc);
        ln_kernel<<<2048, 256, 0, stream>>>(acc, nullptr, mnorm_g, mnorm_b, t_f32, t_bf);
        gemm_kernel<<<512, 256, 0, stream>>>(t_bf, wk_bf, in_proj_b + 256, nullptr, kh, 0);
        gemm_kernel<<<512, 256, 0, stream>>>(t_bf, wv_bf, in_proj_b + 512, nullptr, vv, 0);
        attn_kernel<<<256, 256, 0, stream>>>(qproj, kh, vv, o_bf);
        gemm_kernel<<<512, 256, 0, stream>>>(o_bf, wo_bf, out_proj_b, acc, nullptr, 0);
        ln_kernel<<<2048, 256, 0, stream>>>(acc, t_f32, norm_g, norm_b, fea_f32, fea_bf);
        gemm_kernel<<<512, 256, 0, stream>>>(fea_bf, l1_bf, lin1_b, nullptr, ffa, 1);
        gemm_kernel<<<512, 256, 0, stream>>>(ffa, l2_bf, lin2_b, acc, nullptr, 0);
        ln_kernel<<<2048, 256, 0, stream>>>(acc, fea_f32, normf_g, normf_b, t_f32, nullptr);
        wb_kernel<<<16384, 256, 0, stream>>>(t_f32, words, s, (float*)d_out, featP);
    }
}
